// DeformConvNet_24309514895538
// MI455X (gfx1250) — matrix operand order repaired and checked by reading, not compiled, not run on hardware
//
#include <hip/hip_runtime.h>
#include <cstddef>

// ---------------------------------------------------------------------------
// Types for CDNA5 WMMA (wave32, 16x16x32 bf16 -> f32)
// ---------------------------------------------------------------------------
typedef unsigned short v8us  __attribute__((ext_vector_type(8)));
typedef unsigned short v16us __attribute__((ext_vector_type(16)));
typedef __bf16         v16bf __attribute__((ext_vector_type(16)));
typedef float          v8f   __attribute__((ext_vector_type(8)));

__device__ __forceinline__ float bf2f(unsigned short h) {
    union { unsigned int u; float f; } c;
    c.u = ((unsigned int)h) << 16;
    return c.f;
}
__device__ __forceinline__ unsigned short f2bf(float f) {
    union { float f; unsigned int u; } c;
    c.f = f;
    unsigned int u = c.u;
    unsigned int r = u + 0x7fffu + ((u >> 16) & 1u);   // round-to-nearest-even
    return (unsigned short)(r >> 16);
}

#define TBM 128
#define TBN 256
#define TBK 32
#define LDSTR 40     // skewed LDS row stride (ushorts); keeps 16B alignment
#define DCHUNK 4096  // deform GEMM N-chunk (always fully staged / zero padded)

// ---------------------------------------------------------------------------
// WMMA GEMM:  out[M][outLd](+colOff) = Apk[Mpad][Kpad](bf16) x B + bias
// mode 0: B = implicit im2col of act[Ci][4][Hi][Wi] (3x3 conv, given pad)
// mode 1: B = dense bf16 matrix, TRANSPOSED layout Bmat[n][Kpad], n<DCHUNK
// Apk is zero-padded to Mpad (mult of 128) x Kpad (mult of 32): A staging is
// fully vectorized (global_load_b128 -> ds_store_b128), no guards.
// Each of 8 waves computes a 64x64 sub-tile: 16 v_wmma per K-step.
// ---------------------------------------------------------------------------
__global__ __launch_bounds__(256) void wmma_gemm_kernel(
    const unsigned short* __restrict__ Apk,
    const unsigned short* __restrict__ act,
    const unsigned short* __restrict__ Bmat,
    const float* __restrict__ bias,
    float* __restrict__ out,
    int Mdim, int Kdim, int Kpad, int Nt,
    int mode, int Hi, int Wi, int Ho, int Wo, int pad,
    int outLd, int outColOff)
{
    __shared__ __align__(16) unsigned short As[TBM * LDSTR];
    __shared__ __align__(16) unsigned short Bs[TBN * LDSTR];

    const int tid   = threadIdx.x;
    const int lane  = tid & 31;
    const int wid   = tid >> 5;
    const int waveM = wid >> 2;    // 0..1  (64 rows each)
    const int waveN = wid & 3;     // 0..3  (64 cols each)
    const int l15   = lane & 15;
    const int khalf = lane >> 4;

    const int blkN = blockIdx.x;
    const int blkM = blockIdx.y;

    // This thread's B column (n_local == tid), decomposed once (conv mode).
    const int nG = blkN * TBN + tid;
    const bool nValid = (nG < Nt);
    int bb = 0, oy = 0, ox = 0;
    if (mode == 0) {
        int hw = Ho * Wo;
        bb = nG / hw;
        int r = nG - bb * hw;
        oy = r / Wo;
        ox = r - oy * Wo;
    }

    v8f acc[4][4] = {};

    const int nK = Kpad / TBK;
    for (int kt = 0; kt < nK; ++kt) {
        const int k0 = kt * TBK;

        // ---- stage A tile (128 x 32): 2 x b128 per thread, unguarded ----
        {
            const int idx = tid * 2;
            #pragma unroll
            for (int j = 0; j < 2; ++j) {
                int m    = (idx + j) >> 2;          // 0..127
                int kseg = ((idx + j) & 3) * 8;     // 0,8,16,24
                const unsigned short* src =
                    Apk + (size_t)(blkM * TBM + m) * Kpad + k0 + kseg;
                if (j == 0 && (k0 + TBK) < Kpad)
                    __builtin_prefetch(src + TBK, 0, 3);
                v8us v = *(const v8us*)src;
                *(v8us*)(As + m * LDSTR + kseg) = v;
            }
        }
        // ---- stage B tile (256n x 32k), transposed in LDS: Bs[n][k] ----
        if (mode == 0) {
            #pragma unroll 4
            for (int kl = 0; kl < TBK; ++kl) {
                int kg = k0 + kl;
                unsigned short v = 0;
                if (nValid && kg < Kdim) {
                    int ci = kg / 9;
                    int r9 = kg - ci * 9;
                    int ky = r9 / 3;
                    int kx = r9 - ky * 3;
                    int iy = oy - pad + ky;
                    int ix = ox - pad + kx;
                    if (iy >= 0 && iy < Hi && ix >= 0 && ix < Wi)
                        v = act[(((size_t)ci * 4 + bb) * Hi + iy) * Wi + ix];
                }
                Bs[tid * LDSTR + kl] = v;
            }
        } else {
            // dense transposed B: row = column index, stride Kpad; zero-padded
            const unsigned short* src = Bmat + (size_t)nG * Kpad + k0;
            #pragma unroll
            for (int j = 0; j < 4; ++j) {
                v8us v = *(const v8us*)(src + j * 8);
                *(v8us*)(Bs + tid * LDSTR + j * 8) = v;
            }
        }
        __syncthreads();

        // ---- compute: 4(M) x 4(N) wmma tiles per wave ----
        v16bf afrag[4];
        #pragma unroll
        for (int tm = 0; tm < 4; ++tm) {
            // A 16x32 bf16 layout: lane m=l15; K runs [8*khalf,+8),[16+8*khalf,+8)
            int row = waveM * 64 + tm * 16 + l15;
            v8us lo = *(const v8us*)(As + row * LDSTR + khalf * 8);
            v8us hi = *(const v8us*)(As + row * LDSTR + 16 + khalf * 8);
            v16us m16 = __builtin_shufflevector(lo, hi,
                0,1,2,3,4,5,6,7,8,9,10,11,12,13,14,15);
            afrag[tm] = __builtin_bit_cast(v16bf, m16);
        }
        #pragma unroll
        for (int tn = 0; tn < 4; ++tn) {
            // B 32x16 bf16 layout: lane n=l15; K runs [8*khalf,+8),[16+8*khalf,+8)
            int col = waveN * 64 + tn * 16 + l15;
            v8us lo = *(const v8us*)(Bs + col * LDSTR + khalf * 8);
            v8us hi = *(const v8us*)(Bs + col * LDSTR + 16 + khalf * 8);
            v16us m16 = __builtin_shufflevector(lo, hi,
                0,1,2,3,4,5,6,7,8,9,10,11,12,13,14,15);
            v16bf bfrag = __builtin_bit_cast(v16bf, m16);
            #pragma unroll
            for (int tm = 0; tm < 4; ++tm) {
                acc[tm][tn] = __builtin_amdgcn_wmma_f32_16x16x32_bf16(
                    false, afrag[tm], false, bfrag,
                    (short)0, acc[tm][tn], false, false);
            }
        }
        __syncthreads();
    }

    // ---- epilogue: f32 C/D layout: M = v + 8*khalf, N = l15 ----
    #pragma unroll
    for (int tm = 0; tm < 4; ++tm) {
        #pragma unroll
        for (int tn = 0; tn < 4; ++tn) {
            int colLocal = blkN * TBN + waveN * 64 + tn * 16 + l15;
            if (colLocal >= Nt) continue;
            #pragma unroll
            for (int v = 0; v < 8; ++v) {
                int gr = blkM * TBM + waveM * 64 + tm * 16 + khalf * 8 + v;
                if (gr < Mdim)
                    out[(size_t)gr * outLd + outColOff + colLocal] =
                        acc[tm][tn][v] + bias[gr];
            }
        }
    }
}

// ---------------------------------------------------------------------------
// Small helper kernels
// ---------------------------------------------------------------------------
// fp32 [M][K] -> bf16 zero-padded [Mpad][Kpad]
__global__ void pack_pad_kernel(const float* __restrict__ w,
                                unsigned short* __restrict__ out,
                                int M, int K, int Kpad, int total)
{
    int i = blockIdx.x * 256 + threadIdx.x;
    if (i >= total) return;
    int m = i / Kpad;
    int k = i - m * Kpad;
    out[i] = (m < M && k < K) ? f2bf(w[(size_t)m * K + k]) : (unsigned short)0;
}

__global__ void f32_to_bf16_kernel(const float* __restrict__ in,
                                   unsigned short* __restrict__ out, int n)
{
    int i = blockIdx.x * 256 + threadIdx.x;
    if (i < n) out[i] = f2bf(in[i]);
}

// x [4][C][H][W] fp32 -> act [C][4][H][W] bf16
__global__ void input_cast_kernel(const float* __restrict__ x,
                                  unsigned short* __restrict__ act, int C, int H)
{
    int HW = H * H;
    int tot = C * 4 * HW;
    int i = blockIdx.x * 256 + threadIdx.x;
    if (i >= tot) return;
    int c = i / (4 * HW);
    int r = i - c * 4 * HW;
    int b = r / HW;
    int p = r - b * HW;
    act[i] = f2bf(x[((size_t)b * C + c) * HW + p]);
}

__global__ void bn_stats_kernel(const float* __restrict__ y,
                                float* __restrict__ mean, float* __restrict__ var, int N)
{
    int c = blockIdx.x;
    const float* p = y + (size_t)c * N;
    float s = 0.f, ss = 0.f;
    for (int i = threadIdx.x; i < N; i += 256) { float v = p[i]; s += v; ss += v * v; }
    __shared__ float rs[256], rss[256];
    rs[threadIdx.x] = s; rss[threadIdx.x] = ss;
    __syncthreads();
    for (int st = 128; st > 0; st >>= 1) {
        if (threadIdx.x < st) { rs[threadIdx.x] += rs[threadIdx.x + st];
                                rss[threadIdx.x] += rss[threadIdx.x + st]; }
        __syncthreads();
    }
    if (threadIdx.x == 0) {
        float m = rs[0] / (float)N;
        mean[c] = m;
        var[c] = rss[0] / (float)N - m * m;
    }
}

__global__ void bn_relu_cast_kernel(const float* __restrict__ y,
                                    const float* __restrict__ mean,
                                    const float* __restrict__ var,
                                    const float* __restrict__ g,
                                    const float* __restrict__ be,
                                    unsigned short* __restrict__ outp,
                                    int N, int total)
{
    int i = blockIdx.x * 256 + threadIdx.x;
    if (i >= total) return;
    int c = i / N;
    float inv = rsqrtf(var[c] + 1e-5f);
    float v = g[c] * (y[i] - mean[c]) * inv + be[c];
    outp[i] = f2bf(fmaxf(v, 0.f));
}

// direct 3x3 conv, pad=1: act[C][4][H][H] bf16 -> off[18][4][H][H] fp32
__global__ void offset_conv_kernel(const unsigned short* __restrict__ act,
                                   const float* __restrict__ ow,
                                   const float* __restrict__ ob,
                                   float* __restrict__ off, int C, int H)
{
    int HW = H * H;
    int N = 4 * HW;
    int t = blockIdx.x * blockDim.x + threadIdx.x;
    if (t >= 18 * N) return;
    int oc = t / N;
    int r = t - oc * N;
    int b = r / HW;
    int r2 = r - b * HW;
    int y = r2 / H, x = r2 - (r2 / H) * H;
    float s = ob[oc];
    for (int ci = 0; ci < C; ++ci) {
        const unsigned short* p = act + ((size_t)ci * 4 + b) * HW;
        const float* wv = ow + ((size_t)oc * C + ci) * 9;
        for (int ky = 0; ky < 3; ++ky) {
            int iy = y - 1 + ky;
            if (iy < 0 || iy >= H) continue;
            for (int kx = 0; kx < 3; ++kx) {
                int ix = x - 1 + kx;
                if (ix < 0 || ix >= H) continue;
                s += bf2f(p[iy * H + ix]) * wv[ky * 3 + kx];
            }
        }
    }
    off[t] = s;
}

// bilinear deform sampling for one N-chunk.
// samp layout TRANSPOSED: samp[nl][Kdim], nl in [0,DCHUNK), col = c*9+k2.
// Tail rows (n >= N) are zero-filled so the GEMM B stage is unguarded.
__global__ void deform_sample_kernel(const unsigned short* __restrict__ act,
                                     const float* __restrict__ off,
                                     unsigned short* __restrict__ samp,
                                     int C, int H, int n0)
{
    int t = blockIdx.x * blockDim.x + threadIdx.x;
    if (t >= 9 * DCHUNK) return;
    int k2 = t / DCHUNK;
    int nl = t - k2 * DCHUNK;
    int n = n0 + nl;
    int HW = H * H;
    int N = 4 * HW;
    int Kdim = C * 9;
    unsigned short* row = samp + (size_t)nl * Kdim;
    if (n >= N) {                 // zero-pad tail rows (one worker per row)
        if (k2 == 0)
            for (int k = 0; k < Kdim; ++k) row[k] = 0;
        return;
    }
    int b = n / HW;
    int r = n - b * HW;
    int y = r / H, x = r - (r / H) * H;
    float dy = off[(size_t)(2 * k2) * N + n];
    float dx = off[(size_t)(2 * k2 + 1) * N + n];
    float py = (float)y + (float)(k2 / 3 - 1) + dy;
    float px = (float)x + (float)(k2 % 3 - 1) + dx;
    float y0f = floorf(py), x0f = floorf(px);
    float wy = py - y0f, wx = px - x0f;
    int y0 = (int)y0f, x0 = (int)x0f;
    int y1 = y0 + 1, x1 = x0 + 1;
    bool iy0 = (y0 >= 0) && (y0 < H), iy1 = (y1 >= 0) && (y1 < H);
    bool ix0 = (x0 >= 0) && (x0 < H), ix1 = (x1 >= 0) && (x1 < H);
    float w00 = (iy0 && ix0) ? (1.f - wy) * (1.f - wx) : 0.f;
    float w01 = (iy0 && ix1) ? (1.f - wy) * wx : 0.f;
    float w10 = (iy1 && ix0) ? wy * (1.f - wx) : 0.f;
    float w11 = (iy1 && ix1) ? wy * wx : 0.f;
    int yc0 = min(max(y0, 0), H - 1), yc1 = min(max(y1, 0), H - 1);
    int xc0 = min(max(x0, 0), H - 1), xc1 = min(max(x1, 0), H - 1);
    for (int c = 0; c < C; ++c) {
        const unsigned short* p = act + ((size_t)c * 4 + b) * HW;
        float s = w00 * bf2f(p[yc0 * H + xc0]) + w01 * bf2f(p[yc0 * H + xc1])
                + w10 * bf2f(p[yc1 * H + xc0]) + w11 * bf2f(p[yc1 * H + xc1]);
        row[c * 9 + k2] = f2bf(s);
    }
}

// global average pool: act [C][4][HW] bf16 -> pooled [4][C] fp32
__global__ void pool_kernel(const unsigned short* __restrict__ act,
                            float* __restrict__ pooled, int C, int HW)
{
    int cb = blockIdx.x;          // c*4 + b
    int c = cb >> 2, b = cb & 3;
    const unsigned short* p = act + (size_t)cb * HW;
    float s = 0.f;
    for (int i = threadIdx.x; i < HW; i += 256) s += bf2f(p[i]);
    __shared__ float red[256];
    red[threadIdx.x] = s;
    __syncthreads();
    for (int st = 128; st > 0; st >>= 1) {
        if (threadIdx.x < st) red[threadIdx.x] += red[threadIdx.x + st];
        __syncthreads();
    }
    if (threadIdx.x == 0) pooled[b * C + c] = red[0] / (float)HW;
}

__global__ void fc_softmax_kernel(const float* __restrict__ pooled,
                                  const float* __restrict__ fw,
                                  const float* __restrict__ fb,
                                  float* __restrict__ out, int C, int NC)
{
    int b = blockIdx.x;
    int tid = threadIdx.x;
    __shared__ float logits[1024];
    __shared__ float red[256];
    for (int n = tid; n < NC; n += 256) {
        float s = fb[n];
        const float* pv = pooled + (size_t)b * C;
        const float* wv = fw + (size_t)n * C;
        for (int k = 0; k < C; ++k) s += pv[k] * wv[k];
        logits[n] = s;
    }
    __syncthreads();
    float mx = -3.4e38f;
    for (int n = tid; n < NC; n += 256) mx = fmaxf(mx, logits[n]);
    red[tid] = mx;
    __syncthreads();
    for (int st = 128; st > 0; st >>= 1) {
        if (tid < st) red[tid] = fmaxf(red[tid], red[tid + st]);
        __syncthreads();
    }
    float M = red[0];
    __syncthreads();
    float sm = 0.f;
    for (int n = tid; n < NC; n += 256) sm += expf(logits[n] - M);
    red[tid] = sm;
    __syncthreads();
    for (int st = 128; st > 0; st >>= 1) {
        if (tid < st) red[tid] += red[tid + st];
        __syncthreads();
    }
    float S = red[0];
    for (int n = tid; n < NC; n += 256)
        out[(size_t)b * NC + n] = expf(logits[n] - M) / S;
}

// ---------------------------------------------------------------------------
// Host-side orchestration
// ---------------------------------------------------------------------------
static inline int cdiv(int a, int b) { return (a + b - 1) / b; }
static inline int rup(int a, int b) { return cdiv(a, b) * b; }

static void run_post_act(const float* w, const float* b, const float* g, const float* be,
                         const unsigned short* actIn, unsigned short* actOut,
                         int Ci, int Co, int Hi,
                         float* gout, unsigned short* wpk, float* mean, float* var,
                         hipStream_t s)
{
    int Ho = Hi - 2;
    int N = 4 * Ho * Ho;
    int Kdim = Ci * 9;
    int Kpad = rup(Kdim, TBK);
    int Mpad = rup(Co, TBM);
    int totW = Mpad * Kpad;
    pack_pad_kernel<<<cdiv(totW, 256), 256, 0, s>>>(w, wpk, Co, Kdim, Kpad, totW);
    dim3 grid(cdiv(N, TBN), Mpad / TBM);
    wmma_gemm_kernel<<<grid, 256, 0, s>>>(
        wpk, actIn, (const unsigned short*)nullptr, b, gout,
        Co, Kdim, Kpad, N, /*mode=*/0, Hi, Hi, Ho, Ho, /*pad=*/0, N, 0);
    bn_stats_kernel<<<Co, 256, 0, s>>>(gout, mean, var, N);
    int tot = Co * N;
    bn_relu_cast_kernel<<<cdiv(tot, 256), 256, 0, s>>>(gout, mean, var, g, be,
                                                       actOut, N, tot);
}

static void run_deform(const float* ow, const float* ob, const float* w, const float* b,
                       const unsigned short* actIn, unsigned short* actOut,
                       int C, int H,
                       float* gout, unsigned short* wpk, float* offb,
                       unsigned short* samp, hipStream_t s)
{
    int N = 4 * H * H;
    int totOff = 18 * N;
    offset_conv_kernel<<<cdiv(totOff, 128), 128, 0, s>>>(actIn, ow, ob, offb, C, H);
    int Kdim = C * 9;          // always a multiple of 32 here
    int Mpad = rup(C, TBM);
    int totW = Mpad * Kdim;
    pack_pad_kernel<<<cdiv(totW, 256), 256, 0, s>>>(w, wpk, C, Kdim, Kdim, totW);
    for (int n0 = 0; n0 < N; n0 += DCHUNK) {
        int Nt = (N - n0 < DCHUNK) ? (N - n0) : DCHUNK;
        deform_sample_kernel<<<cdiv(9 * DCHUNK, 256), 256, 0, s>>>(
            actIn, offb, samp, C, H, n0);
        dim3 grid(DCHUNK / TBN, Mpad / TBM);
        wmma_gemm_kernel<<<grid, 256, 0, s>>>(
            wpk, (const unsigned short*)nullptr, samp, b, gout,
            C, Kdim, Kdim, Nt, /*mode=*/1, 0, 0, 0, 0, 0, N, n0);
    }
    int tot = C * N;
    f32_to_bf16_kernel<<<cdiv(tot, 256), 256, 0, s>>>(gout, actOut, tot);
}

extern "C" void kernel_launch(void* const* d_in, const int* in_sizes, int n_in,
                              void* d_out, int out_size, void* d_ws, size_t ws_size,
                              hipStream_t stream)
{
    (void)in_sizes; (void)n_in; (void)out_size; (void)ws_size;
    const float* x = (const float*)d_in[0];
    const float *paw[7], *pab[7], *pag[7], *pabe[7];
    for (int i = 0; i < 7; ++i) {
        paw[i]  = (const float*)d_in[1 + 4 * i];
        pab[i]  = (const float*)d_in[2 + 4 * i];
        pag[i]  = (const float*)d_in[3 + 4 * i];
        pabe[i] = (const float*)d_in[4 + 4 * i];
    }
    const float *dow[3], *dob[3], *dfw[3], *dfb[3];
    for (int j = 0; j < 3; ++j) {
        dow[j] = (const float*)d_in[29 + 4 * j];
        dob[j] = (const float*)d_in[30 + 4 * j];
        dfw[j] = (const float*)d_in[31 + 4 * j];
        dfb[j] = (const float*)d_in[32 + 4 * j];
    }
    const float* fcw = (const float*)d_in[41];
    const float* fcb = (const float*)d_in[42];

    // workspace carve-out
    char* wsp = (char*)d_ws;
    auto carve = [&](size_t bytes) -> void* {
        void* p = (void*)wsp;
        wsp += (bytes + 255) & ~(size_t)255;
        return p;
    };
    const size_t maxActElems = (size_t)512 * 4 * 84 * 84;   // 14.45M
    unsigned short* actA = (unsigned short*)carve(maxActElems * 2);
    unsigned short* actB = (unsigned short*)carve(maxActElems * 2);
    float* gout          = (float*)carve(maxActElems * 4);
    unsigned short* wpk  = (unsigned short*)carve((size_t)512 * 4608 * 2); // Mpad*Kpad max
    float* mean          = (float*)carve(512 * 4);
    float* var           = (float*)carve(512 * 4);
    float* offb          = (float*)carve((size_t)18 * 4 * 94 * 94 * 4);
    unsigned short* samp = (unsigned short*)carve((size_t)DCHUNK * 2304 * 2);
    float* pooled        = (float*)carve(4 * 512 * 4);

    // x -> bf16 [3][4][96][96]
    {
        int tot = 3 * 4 * 96 * 96;
        input_cast_kernel<<<cdiv(tot, 256), 256, 0, stream>>>(x, actA, 3, 96);
    }
    run_post_act(paw[0], pab[0], pag[0], pabe[0], actA, actB,   3,  64, 96,
                 gout, wpk, mean, var, stream);                       // -> 94
    run_deform(dow[0], dob[0], dfw[0], dfb[0], actB, actA, 64, 94,
               gout, wpk, offb, samp, stream);
    run_post_act(paw[1], pab[1], pag[1], pabe[1], actA, actB,  64, 128, 94,
                 gout, wpk, mean, var, stream);                       // -> 92
    run_post_act(paw[2], pab[2], pag[2], pabe[2], actB, actA, 128, 128, 92,
                 gout, wpk, mean, var, stream);                       // -> 90
    run_deform(dow[1], dob[1], dfw[1], dfb[1], actA, actB, 128, 90,
               gout, wpk, offb, samp, stream);
    run_post_act(paw[3], pab[3], pag[3], pabe[3], actB, actA, 128, 256, 90,
                 gout, wpk, mean, var, stream);                       // -> 88
    run_post_act(paw[4], pab[4], pag[4], pabe[4], actA, actB, 256, 256, 88,
                 gout, wpk, mean, var, stream);                       // -> 86
    run_deform(dow[2], dob[2], dfw[2], dfb[2], actB, actA, 256, 86,
               gout, wpk, offb, samp, stream);
    run_post_act(paw[5], pab[5], pag[5], pabe[5], actA, actB, 256, 512, 86,
                 gout, wpk, mean, var, stream);                       // -> 84
    run_post_act(paw[6], pab[6], pag[6], pabe[6], actB, actA, 512, 512, 84,
                 gout, wpk, mean, var, stream);                       // -> 82

    pool_kernel<<<512 * 4, 256, 0, stream>>>(actA, pooled, 512, 82 * 82);
    fc_softmax_kernel<<<4, 256, 0, stream>>>(pooled, fcw, fcb,
                                             (float*)d_out, 512, 1000);
}
